// HGNN_LSTM_42322607735043
// MI455X (gfx1250) — matrix operand order repaired and checked by reading, not compiled, not run on hardware
//
#include <hip/hip_runtime.h>
#include <hip/hip_bf16.h>

// ---------------------------------------------------------------------------
// Problem constants (from setup_inputs)
// ---------------------------------------------------------------------------
#define T_STEPS 336
#define B_SZ    32
#define NH      100
#define NM      150
#define FH      8
#define FM      16
#define HG      64
#define HL      64
#define FUT     24
#define EH      300
#define EM      500

typedef __attribute__((ext_vector_type(16))) __bf16 v16bf;
typedef __attribute__((ext_vector_type(8)))  float  v8f;

union FragU { unsigned short s[16]; int4 i4[2]; v16bf v; };

__device__ __forceinline__ unsigned short f2bf(float v) {
    unsigned int u = __float_as_uint(v);
    u += 0x7FFFu + ((u >> 16) & 1u);        // round-to-nearest-even
    return (unsigned short)(u >> 16);
}
// sigmoid via v_exp_f32 + v_rcp_f32 (no IEEE div sequence).
// exp(-x) -> inf for very negative x, rcp(inf) = 0  => correct saturation.
__device__ __forceinline__ float fsig(float x) {
    return __builtin_amdgcn_rcpf(1.0f + __expf(-x));
}
// tanh(x) = 2*sigmoid(2x) - 1 : one exp + one rcp + one fma, no clamp needed.
__device__ __forceinline__ float ftanh(float x) {
    return 2.0f * __builtin_amdgcn_rcpf(1.0f + __expf(-2.0f * x)) - 1.0f;
}

// ---------------------------------------------------------------------------
// Stage 1: fused hetero GraphConv -> leaky_relu, emitting x in the CDNA5
// bf16 WMMA A-fragment layout:
//   element (m,k) of a 16x32 A tile -> lane = m%16 + 16*((k%16)>=8)
//                                      hw   = (k%8) + 8*(k/16)
// x_frag layout (bf16): [t][n][mt][kc][lane(32)][hw(16)]   (1 KB per fragment)
// One block per graph g = b*T + t.
// ---------------------------------------------------------------------------
__global__ __launch_bounds__(256) void gnn_fuse_kernel(
    const float* __restrict__ xm_all, const float* __restrict__ xh_all,
    const int*   __restrict__ eidx_h, const int*   __restrict__ eidx_m,
    const float* __restrict__ W_rel_m, const float* __restrict__ b_rel_m,
    const float* __restrict__ W_root_m, const float* __restrict__ W_rel_h,
    const float* __restrict__ b_rel_h, const float* __restrict__ W_root_h,
    unsigned short* __restrict__ x_frag)
{
    __shared__ float s_xh[NH * FH];
    __shared__ float s_xm[NM * FM];
    __shared__ float s_ah[NH * FH];
    __shared__ float s_am[NH * FM];
    __shared__ float s_Wrh[HG * FH];    // W_rel_h
    __shared__ float s_Wrt[HG * FH];    // W_root_h + W_root_m (summed once)
    __shared__ float s_Wrm[HG * FM];    // W_rel_m
    __shared__ float s_bias[HG];        // b_rel_h + b_rel_m

    const int g   = blockIdx.x;
    const int b   = g / T_STEPS;
    const int t   = g % T_STEPS;
    const int tid = threadIdx.x;

    const float* xh = xh_all + (size_t)g * (NH * FH);
    const float* xm = xm_all + (size_t)g * (NM * FM);

    for (int i = tid; i < NH * FH; i += 256) { s_xh[i] = xh[i]; s_ah[i] = 0.f; }
    for (int i = tid; i < NM * FM; i += 256) s_xm[i] = xm[i];
    for (int i = tid; i < NH * FM; i += 256) s_am[i] = 0.f;
    for (int i = tid; i < HG * FH; i += 256) {
        s_Wrh[i] = W_rel_h[i];
        s_Wrt[i] = W_root_h[i] + W_root_m[i];
    }
    for (int i = tid; i < HG * FM; i += 256) s_Wrm[i] = W_rel_m[i];
    for (int i = tid; i < HG; i += 256) s_bias[i] = b_rel_h[i] + b_rel_m[i];
    __syncthreads();

    // scatter-add over edges (LDS float atomics -> ds_add_f32)
    for (int e = tid; e < EH; e += 256) {
        int s = eidx_h[e], d = eidx_h[EH + e];
        #pragma unroll
        for (int f = 0; f < FH; ++f)
            atomicAdd(&s_ah[d * FH + f], s_xh[s * FH + f]);
    }
    for (int e = tid; e < EM; e += 256) {
        int s = eidx_m[e], d = eidx_m[EM + e];
        #pragma unroll
        for (int f = 0; f < FM; ++f)
            atomicAdd(&s_am[d * FM + f], s_xm[s * FM + f]);
    }
    __syncthreads();

    const int mt   = b >> 4;
    const int mrow = b & 15;

    for (int idx = tid; idx < NH * HG; idx += 256) {
        const int n = idx >> 6, j = idx & 63;
        float v = s_bias[j];
        #pragma unroll
        for (int k = 0; k < FH; ++k)
            v += s_ah[n * FH + k] * s_Wrh[j * FH + k]
               + s_xh[n * FH + k] * s_Wrt[j * FH + k];
        #pragma unroll
        for (int k = 0; k < FM; ++k)
            v += s_am[n * FM + k] * s_Wrm[j * FM + k];
        v *= 0.5f;
        v = (v > 0.f) ? v : 0.01f * v;          // leaky relu

        // A-fragment address for element (m = b, k = j)
        const int kc   = j >> 5;
        const int kk   = j & 31;
        const int lane = mrow + (((kk & 15) >= 8) ? 16 : 0);
        const int hw   = (kk & 7) + ((kk >= 16) ? 8 : 0);
        const size_t off =
            (((((size_t)t * NH + n) * 2 + mt) * 2 + kc) * 32 + lane) * 16 + hw;
        x_frag[off] = f2bf(v);
    }
}

// ---------------------------------------------------------------------------
// Stage 2: per-node LSTM on the WMMA path.
// grid = NH*2 blocks; block = 128 threads (4 waves). Block owns (node n,
// batch-tile mt). Wave q in [0,4) owns gate tiles nt = q, q+4, q+8, q+12,
// i.e. i/f/g/o for hidden cols [16q,16q+16) and rows [16mt,16mt+16).
// W_ih / W_hh bf16 B-fragments (B layout: lane = n%16 + 16*((k%16)>=8),
// hw = (k%8) + 8*(k/16)) are built once and pinned in VGPRs for all 336 steps.
// h exchanged through a double-buffered LDS A-fragment; c lives in one v8f.
// WMMA chains seed from a zero C (loop-invariant); bias is folded into the
// scalar gate epilogue to avoid per-step accumulator re-splat movs.
// ---------------------------------------------------------------------------
__global__ __launch_bounds__(128) void lstm_head_kernel(
    const unsigned short* __restrict__ x_frag,
    const float* __restrict__ W_ih, const float* __restrict__ W_hh,
    const float* __restrict__ b_ih, const float* __restrict__ b_hh,
    const float* __restrict__ W_lin, const float* __restrict__ b_lin,
    float* __restrict__ out)
{
    __shared__ unsigned short s_h[2][2][32][16]; // [buf][kc][lane][hw] bf16 A-frags
    __shared__ float s_hlast[16][HL];

    const int n    = blockIdx.x >> 1;
    const int mt   = blockIdx.x & 1;
    const int tid  = threadIdx.x;
    const int q    = tid >> 5;          // wave id = gate column group
    const int lane = tid & 31;
    const int jn   = lane & 15;         // N (or M) within 16-tile
    const int khal = (lane >> 4) * 8;   // B-layout: lanes 16-31 carry K+8

    // ---- build weight B-fragments (once) ----------------------------------
    v16bf Bih[4][2], Bhh[4][2];
    float biasv[4];
    #pragma unroll
    for (int gI = 0; gI < 4; ++gI) {
        const int nt = q + 4 * gI;
        const int j  = nt * 16 + jn;                 // gate row in [0,256)
        biasv[gI] = b_ih[n * 256 + j] + b_hh[n * 256 + j];
        #pragma unroll
        for (int kc = 0; kc < 2; ++kc) {
            const float* wi = W_ih + ((size_t)n * 256 + j) * 64 + kc * 32 + khal;
            const float* wh = W_hh + ((size_t)n * 256 + j) * 64 + kc * 32 + khal;
            FragU fi, fh;
            #pragma unroll
            for (int hw = 0; hw < 16; ++hw) {
                fi.s[hw] = f2bf(wi[(hw & 7) + ((hw >> 3) << 4)]);
                fh.s[hw] = f2bf(wh[(hw & 7) + ((hw >> 3) << 4)]);
            }
            Bih[gI][kc] = fi.v;
            Bhh[gI][kc] = fh.v;
        }
    }

    // ---- init h (LDS buf 0) and c (registers) -----------------------------
    for (int i = tid; i < 2 * 32 * 16; i += 128)
        ((unsigned short*)s_h[0])[i] = 0;
    const v8f zero8 = {0.f, 0.f, 0.f, 0.f, 0.f, 0.f, 0.f, 0.f};
    v8f cst = zero8;
    float hout[8];
    #pragma unroll
    for (int r = 0; r < 8; ++r) hout[r] = 0.f;
    __syncthreads();

    // precomputed pieces of the h-fragment store address
    const int kcH   = q >> 1;
    const int hwS   = (jn & 7) + ((q & 1) ? 8 : 0);
    const int lOff  = ((jn >= 8) ? 16 : 0) + ((lane >> 4) << 3); // +8*(lane/16)

    // ---- time loop ---------------------------------------------------------
    for (int t = 0; t < T_STEPS; ++t) {
        const int cur = t & 1, nxt = cur ^ 1;

        v16bf Ax[2], Ah[2];
        #pragma unroll
        for (int kc = 0; kc < 2; ++kc) {
            const unsigned short* xp = x_frag +
                ((((size_t)t * NH + n) * 2 + mt) * 2 + kc) * 512 + (size_t)lane * 16;
            FragU ux, uh;
            ux.i4[0] = *(const int4*)(xp);
            ux.i4[1] = *(const int4*)(xp + 8);
            uh.i4[0] = *(const int4*)(&s_h[cur][kc][lane][0]);
            uh.i4[1] = *(const int4*)(&s_h[cur][kc][lane][8]);
            Ax[kc] = ux.v;
            Ah[kc] = uh.v;
        }
        if (t + 2 < T_STEPS) {
            // Prefetch t+2's full 2 KB fragment region for this (n, mt):
            // 32 lanes x 64 B stride = 2 KB, one cacheline per lane.
            // Locality 3 -> WGP scope: pulls into all cache levels.
            const unsigned short* np = x_frag +
                (((size_t)(t + 2) * NH + n) * 2 + mt) * 1024 + (size_t)lane * 32;
            __builtin_prefetch(np, 0, 3);
        }

        // gates = x_t @ W_ih^T + h @ W_hh^T   (bias added in the epilogue)
        v8f acc[4];
        #pragma unroll
        for (int gI = 0; gI < 4; ++gI) {
            acc[gI] = __builtin_amdgcn_wmma_f32_16x16x32_bf16(
                false, Ax[0], false, Bih[gI][0], (short)0, zero8, false, false);
            acc[gI] = __builtin_amdgcn_wmma_f32_16x16x32_bf16(
                false, Ax[1], false, Bih[gI][1], (short)0, acc[gI], false, false);
            acc[gI] = __builtin_amdgcn_wmma_f32_16x16x32_bf16(
                false, Ah[0], false, Bhh[gI][0], (short)0, acc[gI], false, false);
            acc[gI] = __builtin_amdgcn_wmma_f32_16x16x32_bf16(
                false, Ah[1], false, Bhh[gI][1], (short)0, acc[gI], false, false);
        }

        // gate nonlinearities + state update; acc[0..3] = i,f,g,o for the
        // same (row, hidden-col) block, so everything stays lane-local.
        #pragma unroll
        for (int r = 0; r < 8; ++r) {
            const float iv = fsig(acc[0][r] + biasv[0]);
            const float fv = fsig(acc[1][r] + biasv[1]);
            const float gv = ftanh(acc[2][r] + biasv[2]);
            const float ov = fsig(acc[3][r] + biasv[3]);
            const float cn = fv * cst[r] + iv * gv;
            cst[r] = cn;
            const float hv = ov * ftanh(cn);
            hout[r] = hv;
            // store h element (m = r + 8*(lane/16), k = 16q + jn) as bf16
            // into the A-fragment for the next step.
            s_h[nxt][kcH][r + lOff][hwS] = f2bf(hv);
        }
        __syncthreads();   // double-buffered: one barrier per step
    }

    // ---- final projection: pred = h_last @ W_lin^T + b_lin, leaky ---------
    #pragma unroll
    for (int r = 0; r < 8; ++r)
        s_hlast[r + ((lane >> 4) << 3)][q * 16 + jn] = hout[r];
    __syncthreads();

    for (int idx = tid; idx < 16 * FUT; idx += 128) {
        const int br = idx / FUT;               // row within this mt tile
        const int fu = idx % FUT;
        float v = b_lin[fu];
        #pragma unroll 8
        for (int k = 0; k < HL; ++k)
            v += s_hlast[br][k] * W_lin[fu * HL + k];
        v = (v > 0.f) ? v : 0.01f * v;
        const int bb = mt * 16 + br;
        out[((size_t)bb * NH + n) * FUT + fu] = v;
    }
}

// ---------------------------------------------------------------------------
// Launch
// ---------------------------------------------------------------------------
extern "C" void kernel_launch(void* const* d_in, const int* in_sizes, int n_in,
                              void* d_out, int out_size, void* d_ws, size_t ws_size,
                              hipStream_t stream) {
    const float* data_meteo = (const float*)d_in[0];
    const float* data_hydro = (const float*)d_in[1];
    const int*   eidx_h     = (const int*)d_in[2];
    const int*   eidx_m     = (const int*)d_in[3];
    const float* W_rel_m    = (const float*)d_in[4];
    const float* b_rel_m    = (const float*)d_in[5];
    const float* W_root_m   = (const float*)d_in[6];
    const float* W_rel_h    = (const float*)d_in[7];
    const float* b_rel_h    = (const float*)d_in[8];
    const float* W_root_h   = (const float*)d_in[9];
    const float* W_ih       = (const float*)d_in[10];
    const float* W_hh       = (const float*)d_in[11];
    const float* b_ih       = (const float*)d_in[12];
    const float* b_hh       = (const float*)d_in[13];
    const float* W_lin      = (const float*)d_in[14];
    const float* b_lin      = (const float*)d_in[15];

    // x fragments: T*NH*2*2*512 bf16 = 137.6 MB of workspace
    unsigned short* x_frag = (unsigned short*)d_ws;

    gnn_fuse_kernel<<<B_SZ * T_STEPS, 256, 0, stream>>>(
        data_meteo, data_hydro, eidx_h, eidx_m,
        W_rel_m, b_rel_m, W_root_m, W_rel_h, b_rel_h, W_root_h, x_frag);

    lstm_head_kernel<<<NH * 2, 128, 0, stream>>>(
        x_frag, W_ih, W_hh, b_ih, b_hh, W_lin, b_lin, (float*)d_out);
}
